// MultiHeadSelfAttention_84645215470162
// MI455X (gfx1250) — matrix operand order repaired and checked by reading, not compiled, not run on hardware
//
#include <hip/hip_runtime.h>
#include <hip/hip_bf16.h>
#include <stdint.h>

// ---------------------------------------------------------------------------
// MHA for MI455X (gfx1250): bf16 WMMA, f32 accum, flash-attention with
// async global->LDS staging of K/V tiles (ASYNCcnt path).
// ---------------------------------------------------------------------------

typedef __attribute__((ext_vector_type(16))) __bf16 v16bf;
typedef __attribute__((ext_vector_type(8)))  float  v8f;
typedef __hip_bfloat16 bf16_t;

#define SEQ    2048
#define BATCH  4
#define HEADS  16
#define HD     64
#define DMODEL 1024

union FragU { v16bf v; uint4 u[2]; };

__device__ __forceinline__ v8f wmma_bf16(v16bf a, v16bf b, v8f c) {
    // 8 args: (neg_a, A, neg_b, B, c_mod, C, reuse_a, reuse_b)
    return __builtin_amdgcn_wmma_f32_16x16x32_bf16(
        false, a, false, b, (short)0, c, false, false);
}

// A-matrix 16x32 bf16 fragment (ISA 7.12.2): lane L holds row M=L&15.
// Lanes 0-15:  K 0..7 (bytes [0,16)) and K 16..23 (bytes [32,48))
// Lanes 16-31: K 8..15 (bytes [16,32)) and K 24..31 (bytes [48,64))
__device__ __forceinline__ v16bf load_a_frag(const bf16_t* base, int rowStride, int lane) {
    const char* p = (const char*)(base + (size_t)(lane & 15) * rowStride)
                    + ((lane >> 4) & 1) * 16;
    FragU f;
    f.u[0] = *(const uint4*)(p);
    f.u[1] = *(const uint4*)(p + 32);
    return f.v;
}

// B-matrix 32x16 bf16 fragment, loaded from B^T (N rows x K cols, row-major):
// lane L holds column N=L&15; same K order as A: half h holds K 8h..8h+7 and 16+8h..16+8h+7.
__device__ __forceinline__ v16bf load_b_frag(const bf16_t* base, int rowStride, int lane) {
    const char* p = (const char*)(base + (size_t)(lane & 15) * rowStride)
                    + ((lane >> 4) & 1) * 16;
    FragU f;
    f.u[0] = *(const uint4*)(p);
    f.u[1] = *(const uint4*)(p + 32);
    return f.v;
}

// Async global -> LDS copy, 16B per lane. VDST = per-lane LDS byte address,
// VADDR = 64-bit global address (GV mode). Tracked with ASYNCcnt.
__device__ __forceinline__ void async_copy_b128(unsigned lds_addr, const void* gaddr) {
    asm volatile("global_load_async_to_lds_b128 %0, %1, off"
                 :: "v"(lds_addr), "v"(gaddr) : "memory");
}
__device__ __forceinline__ void wait_asynccnt0() {
    asm volatile("s_wait_asynccnt 0x0" ::: "memory");
}

// ---------------------------------------------------------------------------
// f32 -> bf16 convert (with optional scale folded in)
// ---------------------------------------------------------------------------
__global__ void cvt_bf16_kernel(const float* __restrict__ in,
                                bf16_t* __restrict__ out, size_t n, float scale) {
    size_t i = (size_t)blockIdx.x * blockDim.x + threadIdx.x;
    if (i < n) out[i] = __float2bfloat16(in[i] * scale);
}

// ---------------------------------------------------------------------------
// NT GEMM: C[m][n] = sum_k A[m][k] * Bt[n][k], bf16 in, f32 accum.
// Block = 8 waves covering 128(M) x 128(N); each wave does 32(M) x 64(N),
// reusing every B fragment across two M sub-tiles: 12 b128 loads / 8 WMMAs.
// MODE 0: bf16 out, row-major ldc=N        (used for V^T)
// MODE 1: bf16 out, [head][s][64] layout   (used for Q, K)
// MODE 2: f32 out,  row-major ldc=N        (final projection -> d_out)
// ---------------------------------------------------------------------------
template <int MODE>
__global__ void __launch_bounds__(256)
gemm_nt_kernel(const bf16_t* __restrict__ A, const bf16_t* __restrict__ Bt,
               void* __restrict__ Cout, int M, int N, int Kd, int Sdim,
               size_t strideA, size_t strideB, size_t strideC) {
    int wave = threadIdx.x >> 5;
    int lane = threadIdx.x & 31;
    int bz   = blockIdx.z;
    const bf16_t* Ab = A + (size_t)bz * strideA;
    const bf16_t* Bb = Bt + (size_t)bz * strideB;
    int m0 = blockIdx.x * 128 + (wave & 3) * 32;   // wave owns rows [m0, m0+32)
    int n0 = blockIdx.y * 128 + (wave >> 2) * 64;  // wave owns cols [n0, n0+64)

    v8f acc[2][4] = {{v8f{}, v8f{}, v8f{}, v8f{}},
                     {v8f{}, v8f{}, v8f{}, v8f{}}};
    for (int k = 0; k < Kd; k += 32) {
        // prefetch next K slice while this one computes (global_prefetch_b8)
        if (k + 32 < Kd) {
            __builtin_prefetch(Ab + (size_t)(m0 + (lane & 15)) * Kd + k + 32, 0, 1);
            __builtin_prefetch(Bb + (size_t)(n0 + (lane & 15)) * Kd + k + 32, 0, 1);
        }
        v16bf a0 = load_a_frag(Ab + (size_t)m0 * Kd + k,        Kd, lane);
        v16bf a1 = load_a_frag(Ab + (size_t)(m0 + 16) * Kd + k, Kd, lane);
#pragma unroll
        for (int t = 0; t < 4; ++t) {
            v16bf b = load_b_frag(Bb + (size_t)(n0 + t * 16) * Kd + k, Kd, lane);
            acc[0][t] = wmma_bf16(a0, b, acc[0][t]);
            acc[1][t] = wmma_bf16(a1, b, acc[1][t]);
        }
    }

    int half = lane >> 4, nc = lane & 15;
#pragma unroll
    for (int ms = 0; ms < 2; ++ms) {
#pragma unroll
        for (int t = 0; t < 4; ++t) {
#pragma unroll
            for (int i = 0; i < 8; ++i) {
                int mm = m0 + ms * 16 + i + 8 * half;  // C-layout row
                int nn = n0 + t * 16 + nc;             // C-layout col
                float v = acc[ms][t][i];
                if (MODE == 0) {
                    bf16_t* C = (bf16_t*)Cout + (size_t)bz * strideC;
                    C[(size_t)mm * N + nn] = __float2bfloat16(v);
                } else if (MODE == 1) {
                    bf16_t* C = (bf16_t*)Cout + (size_t)bz * strideC;
                    C[(size_t)(nn >> 6) * Sdim * HD + (size_t)mm * HD + (nn & 63)] =
                        __float2bfloat16(v);
                } else {
                    float* C = (float*)Cout;
                    C[(size_t)mm * N + nn] = v;
                }
            }
        }
    }
}

// ---------------------------------------------------------------------------
// Flash attention: grid (S/128, B*H), 256 threads (8 waves).
// Wave w owns 16 query rows; 32 keys per iteration; online softmax.
// K (32x64) and V^T (64x32) tiles are async-staged into LDS once per block
// per iteration (8x traffic reduction vs per-wave fragment loads).
// Q pre-scaled by 1/sqrt(64). Output bf16 [B][S][H*64].
// ---------------------------------------------------------------------------
__global__ void __launch_bounds__(256)
flash_attn_kernel(const bf16_t* __restrict__ Q, const bf16_t* __restrict__ K,
                  const bf16_t* __restrict__ Vt, bf16_t* __restrict__ AO) {
    __shared__ bf16_t lds_k[32][72];      // K tile: rows=key, cols=hd   (144B rows)
    __shared__ bf16_t lds_v[64][40];      // V^T tile: rows=v, cols=key  (80B rows)
    __shared__ bf16_t lds_p[8][16][40];   // per-wave P tile

    int tid  = threadIdx.x;
    int wave = tid >> 5;
    int lane = tid & 31;
    int bh   = blockIdx.y;
    int b    = bh >> 4;          // / HEADS
    int h    = bh & 15;          // % HEADS
    int qblock = blockIdx.x * 128;
    int qbase  = qblock + wave * 16;

    const bf16_t* Qh = Q  + (size_t)bh * SEQ * HD;   // [S][64]
    const bf16_t* Kh = K  + (size_t)bh * SEQ * HD;   // [S][64]
    const bf16_t* Vh = Vt + (size_t)bh * HD * SEQ;   // [64][S]  (V transposed)

    v16bf aq0 = load_a_frag(Qh + (size_t)qbase * HD + 0,  HD, lane);
    v16bf aq1 = load_a_frag(Qh + (size_t)qbase * HD + 32, HD, lane);

    // cooperative copy assignment: 256 threads x 16B = one 4KB tile each
    int krow = tid >> 3, kcol = (tid & 7) * 8;   // K: 32 rows x 8 chunks
    int vrow = tid >> 2, vcol = (tid & 3) * 8;   // V: 64 rows x 4 chunks
    unsigned ldsk_addr = (unsigned)(uintptr_t)&lds_k[krow][kcol];
    unsigned ldsv_addr = (unsigned)(uintptr_t)&lds_v[vrow][vcol];

    v8f o[4] = {v8f{}, v8f{}, v8f{}, v8f{}};
    float rm[8], rs[8];
#pragma unroll
    for (int i = 0; i < 8; ++i) { rm[i] = -3.0e38f; rs[i] = 0.0f; }

    int half = lane >> 4, nc = lane & 15;
    int kend = qblock + 128;                  // causal bound for this block

    for (int kb = 0; kb < kend; kb += 32) {
        // ---- async-stage K/V tiles for this iteration into LDS
        async_copy_b128(ldsk_addr, Kh + (size_t)(kb + krow) * HD + kcol);
        async_copy_b128(ldsv_addr, Vh + (size_t)vrow * SEQ + kb + vcol);
        wait_asynccnt0();
        __syncthreads();

        // ---- scores S = Q . K^T  (two 16x16 N tiles, K=64 via 2 chained wmma)
        v8f s0 = {}, s1 = {};
        {
            v16bf bk;
            bk = load_b_frag(&lds_k[0][0],   72, lane);
            s0 = wmma_bf16(aq0, bk, s0);
            bk = load_b_frag(&lds_k[0][32],  72, lane);
            s0 = wmma_bf16(aq1, bk, s0);
            bk = load_b_frag(&lds_k[16][0],  72, lane);
            s1 = wmma_bf16(aq0, bk, s1);
            bk = load_b_frag(&lds_k[16][32], 72, lane);
            s1 = wmma_bf16(aq1, bk, s1);
        }

        // ---- causal mask (finite large-negative so exp -> 0, never NaN)
        int col0 = kb + nc, col1 = col0 + 16;
#pragma unroll
        for (int i = 0; i < 8; ++i) {
            int qr = qbase + i + 8 * half;
            if (col0 > qr) s0[i] = -3.0e38f;
            if (col1 > qr) s1[i] = -3.0e38f;
        }

        // ---- online softmax (row reductions across the 16-lane half)
#pragma unroll
        for (int i = 0; i < 8; ++i) {
            float t = fmaxf(s0[i], s1[i]);
            t = fmaxf(t, __shfl_xor(t, 1));
            t = fmaxf(t, __shfl_xor(t, 2));
            t = fmaxf(t, __shfl_xor(t, 4));
            t = fmaxf(t, __shfl_xor(t, 8));
            float nm    = fmaxf(rm[i], t);
            float alpha = __expf(rm[i] - nm);
            rm[i] = nm;
            float p0 = __expf(s0[i] - nm);
            float p1 = __expf(s1[i] - nm);
            s0[i] = p0; s1[i] = p1;
            float sum = p0 + p1;
            sum += __shfl_xor(sum, 1);
            sum += __shfl_xor(sum, 2);
            sum += __shfl_xor(sum, 4);
            sum += __shfl_xor(sum, 8);
            rs[i] = rs[i] * alpha + sum;
#pragma unroll
            for (int t4 = 0; t4 < 4; ++t4) o[t4][i] *= alpha;
        }

        // ---- C-layout -> A-layout for P via LDS roundtrip
#pragma unroll
        for (int i = 0; i < 8; ++i) {
            int r = i + 8 * half;
            lds_p[wave][r][nc]      = __float2bfloat16(s0[i]);
            lds_p[wave][r][16 + nc] = __float2bfloat16(s1[i]);
        }
        __syncthreads();
        v16bf ap = load_a_frag(&lds_p[wave][0][0], 40, lane);

        // ---- O += P . V   (B operand from staged V^T rows)
#pragma unroll
        for (int t4 = 0; t4 < 4; ++t4) {
            v16bf bv = load_b_frag(&lds_v[t4 * 16][0], 40, lane);
            o[t4] = wmma_bf16(ap, bv, o[t4]);
        }
        __syncthreads();   // protect lds_k/lds_v before next async fill
    }

    // ---- normalize and store concat-head output [B][S][1024] bf16
#pragma unroll
    for (int t4 = 0; t4 < 4; ++t4) {
#pragma unroll
        for (int i = 0; i < 8; ++i) {
            int qr  = qbase + i + 8 * half;
            int col = h * 64 + t4 * 16 + nc;
            AO[((size_t)b * SEQ + qr) * DMODEL + col] =
                __float2bfloat16(o[t4][i] / rs[i]);
        }
    }
}

// ---------------------------------------------------------------------------
// Host launcher
// ---------------------------------------------------------------------------
extern "C" void kernel_launch(void* const* d_in, const int* in_sizes, int n_in,
                              void* d_out, int out_size, void* d_ws, size_t ws_size,
                              hipStream_t stream) {
    (void)in_sizes; (void)n_in; (void)out_size; (void)ws_size;
    const float* x  = (const float*)d_in[0];
    const float* Wq = (const float*)d_in[1];
    const float* Wk = (const float*)d_in[2];
    const float* Wv = (const float*)d_in[3];
    const float* Wo = (const float*)d_in[4];

    const size_t NX = (size_t)BATCH * SEQ * DMODEL;   // 8,388,608
    const size_t NW = (size_t)DMODEL * DMODEL;        // 1,048,576

    // workspace layout (bf16): xb | wq | wk | wv | wo | Q | K | Vt | AO  (~92 MB)
    bf16_t* xb  = (bf16_t*)d_ws;
    bf16_t* wqb = xb  + NX;
    bf16_t* wkb = wqb + NW;
    bf16_t* wvb = wkb + NW;
    bf16_t* wob = wvb + NW;
    bf16_t* Qb  = wob + NW;
    bf16_t* Kb  = Qb  + NX;
    bf16_t* Vtb = Kb  + NX;
    bf16_t* AO  = Vtb + NX;

    const float qscale = 0.125f;  // 1/sqrt(64) folded into W_Q

    cvt_bf16_kernel<<<dim3((unsigned)((NX + 255) / 256)), 256, 0, stream>>>(x,  xb,  NX, 1.0f);
    cvt_bf16_kernel<<<dim3((unsigned)((NW + 255) / 256)), 256, 0, stream>>>(Wq, wqb, NW, qscale);
    cvt_bf16_kernel<<<dim3((unsigned)((NW + 255) / 256)), 256, 0, stream>>>(Wk, wkb, NW, 1.0f);
    cvt_bf16_kernel<<<dim3((unsigned)((NW + 255) / 256)), 256, 0, stream>>>(Wv, wvb, NW, 1.0f);
    cvt_bf16_kernel<<<dim3((unsigned)((NW + 255) / 256)), 256, 0, stream>>>(Wo, wob, NW, 1.0f);

    dim3 blk(256);

    // Q = x . Wq^T  -> [B][H][S][64]
    gemm_nt_kernel<1><<<dim3(SEQ / 128, DMODEL / 128, BATCH), blk, 0, stream>>>(
        xb, wqb, Qb, SEQ, DMODEL, DMODEL, SEQ,
        (size_t)SEQ * DMODEL, 0, (size_t)HEADS * SEQ * HD);
    // K = x . Wk^T  -> [B][H][S][64]
    gemm_nt_kernel<1><<<dim3(SEQ / 128, DMODEL / 128, BATCH), blk, 0, stream>>>(
        xb, wkb, Kb, SEQ, DMODEL, DMODEL, SEQ,
        (size_t)SEQ * DMODEL, 0, (size_t)HEADS * SEQ * HD);
    // V^T = Wv . x^T -> [B][H*64][S]   (rows = head*64+v, contiguous in s)
    gemm_nt_kernel<0><<<dim3(DMODEL / 128, SEQ / 128, BATCH), blk, 0, stream>>>(
        wvb, xb, Vtb, DMODEL, SEQ, DMODEL, 0,
        0, (size_t)SEQ * DMODEL, (size_t)DMODEL * SEQ);

    // causal flash attention -> AO [B][S][1024] bf16
    flash_attn_kernel<<<dim3(SEQ / 128, BATCH * HEADS), blk, 0, stream>>>(Qb, Kb, Vtb, AO);

    // out = AO . Wo^T -> f32 [B*S][1024]
    gemm_nt_kernel<2><<<dim3((BATCH * SEQ) / 128, DMODEL / 128, 1), blk, 0, stream>>>(
        AO, wob, d_out, BATCH * SEQ, DMODEL, DMODEL, 0, 0, 0, 0);
}
